// CausalSelfAttentionMLA_86870008529342
// MI455X (gfx1250) — matrix operand order repaired and checked by reading, not compiled, not run on hardware
//
#include <hip/hip_runtime.h>

// ---------------------------------------------------------------------------
// CDNA5 (gfx1250) MLA causal attention — bf16 WMMA + async global->LDS staging.
// Pipeline:
//   xb   = bf16(x)                 ; Wlt/Wdt/Wpt = bf16(transpose(W_*))
//   kv   = xb @ Wlt^T + b_latent   (fp32, 4096x128)
//   q    = xb @ Wdt^T + b_d        (fp32, 4096x2048)
//   krot/kv16 = rope/convert(kv)   (bf16) ;  qrot = rope(q) (bf16)
//   y    = flash_attn(qrot, krot, kv16)   (bf16)
//   out  = y @ Wpt^T + b_proj      (fp32)
// ---------------------------------------------------------------------------

typedef __attribute__((ext_vector_type(16))) __bf16 v16bf;
typedef __attribute__((ext_vector_type(8)))  __bf16 v8bf;
typedef __attribute__((ext_vector_type(2)))  __bf16 v2bf;
typedef __attribute__((ext_vector_type(8)))  float  v8f;

// ---- CDNA5 async global->LDS copy (ASYNCcnt) ------------------------------
__device__ __forceinline__ void async_copy_b128(void* lds_dst, const void* gsrc) {
  unsigned lds = (unsigned)(unsigned long long)lds_dst;   // low 32b = LDS offset
  unsigned long long ga = (unsigned long long)gsrc;
  asm volatile("global_load_async_to_lds_b128 %0, %1, off"
               :: "v"(lds), "v"(ga) : "memory");
}
__device__ __forceinline__ void s_wait_async0() {
  asm volatile("s_wait_asynccnt 0" ::: "memory");
}

// Build a 16x32 A-fragment / 32x16 B-fragment from two contiguous 16B chunks.
__device__ __forceinline__ v16bf ld_frag(const __bf16* p0, const __bf16* p1) {
  v8bf lo = *(const v8bf*)p0;
  v8bf hi = *(const v8bf*)p1;
  v16bf r;
#pragma unroll
  for (int e = 0; e < 8; ++e) { r[e] = lo[e]; r[8 + e] = hi[e]; }
  return r;
}

__device__ __forceinline__ v8f wmma_bf16(v16bf a, v16bf b, v8f c) {
  return __builtin_amdgcn_wmma_f32_16x16x32_bf16(false, a, false, b,
                                                 (short)0, c, false, false);
}

// ---------------------------------------------------------------------------
// fp32 -> bf16 bulk convert (8 elems / thread).
// ---------------------------------------------------------------------------
__global__ __launch_bounds__(256) void convert_bf16_kernel(
    const float* __restrict__ in, __bf16* __restrict__ out, int nchunks)
{
  int i = blockIdx.x * 256 + threadIdx.x;
  if (i >= nchunks) return;
  const float4* p = (const float4*)(in + (size_t)i * 8);
  float4 a = p[0], b = p[1];
  v8bf o;
  o[0] = (__bf16)a.x; o[1] = (__bf16)a.y; o[2] = (__bf16)a.z; o[3] = (__bf16)a.w;
  o[4] = (__bf16)b.x; o[5] = (__bf16)b.y; o[6] = (__bf16)b.z; o[7] = (__bf16)b.w;
  *(v8bf*)(out + (size_t)i * 8) = o;
}

// ---------------------------------------------------------------------------
// fp32 (R x C) -> bf16 transpose (C x R).  32x32 LDS tiles.
// ---------------------------------------------------------------------------
__global__ __launch_bounds__(256) void transpose_to_bf16(
    const float* __restrict__ in, __bf16* __restrict__ out, int R, int Ccol)
{
  __shared__ float tile[32][33];
  const int tx = threadIdx.x & 31, ty = threadIdx.x >> 5;
  const int bx = blockIdx.x * 32;  // column base of `in`
  const int by = blockIdx.y * 32;  // row base of `in`
#pragma unroll
  for (int j = 0; j < 4; ++j)
    tile[ty + j * 8][tx] = in[(size_t)(by + ty + j * 8) * Ccol + bx + tx];
  __syncthreads();
#pragma unroll
  for (int j = 0; j < 4; ++j)
    out[(size_t)(bx + ty + j * 8) * R + by + tx] = (__bf16)tile[tx][ty + j * 8];
}

// ---------------------------------------------------------------------------
// GEMM: C(MxN,f32) = A(MxK,bf16) @ Bt(NxK,bf16)^T + bias.
// 256 threads / 8 waves, 128x128 tile, K-step 32, async LDS staging.
// Each wave owns a 16-row strip and all 8 column subtiles (8 wmma / K-tile).
// ---------------------------------------------------------------------------
#define GM 128
#define GN 128
#define GK 32

__global__ __launch_bounds__(256) void gemm_bf16t_wmma(
    const __bf16* __restrict__ A, const __bf16* __restrict__ Bt,
    const float* __restrict__ bias, float* __restrict__ C,
    int M, int N, int K)
{
  __shared__ __attribute__((aligned(16))) __bf16 As[GM][GK + 8];  // stride 80B
  __shared__ __attribute__((aligned(16))) __bf16 Bs[GN][GK + 8];

  const int tid = threadIdx.x, wave = tid >> 5, lane = tid & 31;
  const int m0 = blockIdx.y * GM, n0 = blockIdx.x * GN;

  v8f acc[8] = {};

  const int rA  = wave * 16 + (lane & 15);
  const int kbA = (lane < 16) ? 0 : 8;
  const int kbB = (lane < 16) ? 0 : 8;

  for (int k0 = 0; k0 < K; k0 += GK) {
    // Stage both tiles: 4096 bf16 each, 16B chunks, 2 per thread per matrix.
#pragma unroll
    for (int j = 0; j < 2; ++j) {
      int chunk = j * 256 + tid;          // 512 chunks of 8 elems
      int rr = chunk >> 2;                // 4 chunks per 32-wide row
      int cc = (chunk & 3) * 8;
      async_copy_b128(&As[rr][cc], &A[(size_t)(m0 + rr) * K + k0 + cc]);
      async_copy_b128(&Bs[rr][cc], &Bt[(size_t)(n0 + rr) * K + k0 + cc]);
    }
    s_wait_async0();
    __syncthreads();

    v16bf a = ld_frag(&As[rA][kbA], &As[rA][16 + kbA]);
#pragma unroll
    for (int ci = 0; ci < 8; ++ci) {
      int nB = ci * 16 + (lane & 15);
      v16bf b = ld_frag(&Bs[nB][kbB], &Bs[nB][16 + kbB]);
      acc[ci] = wmma_bf16(a, b, acc[ci]);
    }
    __syncthreads();
  }

  const int rowBase = m0 + wave * 16 + ((lane >> 4) << 3);
#pragma unroll
  for (int ci = 0; ci < 8; ++ci) {
    int col = n0 + ci * 16 + (lane & 15);
    float bb = bias[col];
#pragma unroll
    for (int i = 0; i < 8; ++i)
      C[(size_t)(rowBase + i) * N + col] = acc[ci][i] + bb;
  }
}

// ---------------------------------------------------------------------------
// RoPE (interleaved) fp32 -> bf16; optionally also emit unrotated bf16 copy.
// Rows are (total_rows) vectors of 128; t = (row / heads) % T.
// ---------------------------------------------------------------------------
__global__ __launch_bounds__(256) void rope_to_bf16(
    const float* __restrict__ in, __bf16* __restrict__ rot,
    __bf16* __restrict__ plain, int total_rows, int heads, int T)
{
  int pair = blockIdx.x * 256 + threadIdx.x;    // over total_rows * 64
  int row = pair >> 6;
  if (row >= total_rows) return;
  int j = pair & 63;
  int t = (row / heads) % T;
  float inv_freq = __powf(10000.0f, -(float)j * (1.0f / 64.0f));
  float ang = (float)t * inv_freq;
  float c = __cosf(ang), s = __sinf(ang);
  size_t base = (size_t)row * 128 + j * 2;
  float e = in[base], o = in[base + 1];
  v2bf r; r[0] = (__bf16)(e * c - o * s); r[1] = (__bf16)(o * c + e * s);
  *(v2bf*)&rot[base] = r;
  if (plain) {
    v2bf p; p[0] = (__bf16)e; p[1] = (__bf16)o;
    *(v2bf*)&plain[base] = p;
  }
}

// ---------------------------------------------------------------------------
// Flash attention: per (b,h), 64-query block, stream 32-key blocks causally.
// All inputs bf16; output y bf16 (B,T,H*128).
// ---------------------------------------------------------------------------
#define FBQ 64
#define FBK 32
#define FD  128

__global__ __launch_bounds__(256) void flash_mla_wmma(
    const __bf16* __restrict__ qrot,  // (B,T,H,128)
    const __bf16* __restrict__ krot,  // (B,T,128)
    const __bf16* __restrict__ kv16,  // (B,T,128)
    __bf16* __restrict__ y,           // (B,T,H*128)
    int B, int T)
{
  __shared__ __attribute__((aligned(16))) __bf16 Qs[FBQ][FD + 8];
  __shared__ __attribute__((aligned(16))) __bf16 Ks[FBK][FD + 8];
  __shared__ __attribute__((aligned(16))) __bf16 Vst[FD][FBK + 8];  // transposed
  __shared__ __attribute__((aligned(16))) float  Ss[FBQ][FBK + 8];
  __shared__ __attribute__((aligned(16))) __bf16 Ps[FBQ][FBK + 8];
  __shared__ float row_m[FBQ], row_l[FBQ], row_c[FBQ];

  const int tid = threadIdx.x, wave = tid >> 5, lane = tid & 31;
  const int bh = blockIdx.x;
  const int b = bh >> 4, h = bh & 15;
  const int qy = blockIdx.y;
  const int qb = qy * FBQ;

  // Async-stage the 64x128 Q block (bf16, 16B chunks, 4 per thread).
#pragma unroll
  for (int j = 0; j < 4; ++j) {
    int chunk = j * 256 + tid;        // 1024 chunks
    int rr = chunk >> 4;              // 16 chunks per 128-wide row
    int d0 = (chunk & 15) * 8;
    async_copy_b128(&Qs[rr][d0],
                    &qrot[(((size_t)b * T + (qb + rr)) * 16 + h) * 128 + d0]);
  }
  if (tid < FBQ) { row_m[tid] = -__builtin_inff(); row_l[tid] = 0.f; }

  const int mi  = wave & 3;
  const int ni  = wave >> 2;
  const int oci = (wave >> 2) * 4;
  const int rA  = mi * 16 + (lane & 15);
  const int nB  = ni * 16 + (lane & 15);
  const int kbA = (lane < 16) ? 0 : 8;
  const int kbB = (lane < 16) ? 0 : 8;
  const int m_base = mi * 16 + ((lane >> 4) << 3);

  v8f Oacc[4] = {};
  const float scale = 0.08838834764831845f;  // 1/sqrt(128)

  const int nkb = 2 * qy + 2;
  for (int kb = 0; kb < nkb; ++kb) {
    __syncthreads();  // previous iteration's LDS reads complete
    const int s0 = kb * FBK;
    // K block: async copy (512 chunks, 2/thread).  V block: transposed scatter.
#pragma unroll
    for (int j = 0; j < 2; ++j) {
      int chunk = j * 256 + tid;
      int rr = chunk >> 4, d0 = (chunk & 15) * 8;
      size_t g = (size_t)b * T + (s0 + rr);
      async_copy_b128(&Ks[rr][d0], &krot[g * 128 + d0]);
      v8bf v = *(const v8bf*)&kv16[g * 128 + d0];
#pragma unroll
      for (int e = 0; e < 8; ++e) Vst[d0 + e][rr] = v[e];
    }
    s_wait_async0();
    __syncthreads();

    // S = Q @ K^T  (per-wave 16x16 subtile, 4 k-chunks over d=128)
    v8f sacc = {};
#pragma unroll
    for (int kc = 0; kc < 4; ++kc) {
      v16bf a  = ld_frag(&Qs[rA][kc * 32 + kbA], &Qs[rA][kc * 32 + 16 + kbA]);
      v16bf bb = ld_frag(&Ks[nB][kc * 32 + kbB], &Ks[nB][kc * 32 + 16 + kbB]);
      sacc = wmma_bf16(a, bb, sacc);
    }
    // Scale + causal mask, spill S subtile to LDS.
    {
      const int ncol = ni * 16 + (lane & 15);
      const int s_glob = s0 + ncol;
#pragma unroll
      for (int i = 0; i < 8; ++i) {
        float v = sacc[i] * scale;
        if (s_glob > qb + m_base + i) v = -__builtin_inff();
        Ss[m_base + i][ncol] = v;
      }
    }
    __syncthreads();

    // Online softmax: one thread per query row.
    if (tid < FBQ) {
      float mo = row_m[tid];
      float mx = mo;
#pragma unroll
      for (int jj = 0; jj < FBK; ++jj) mx = fmaxf(mx, Ss[tid][jj]);
      float c = 1.0f, ladd = 0.f;
      if (mx == -__builtin_inff()) {
#pragma unroll
        for (int jj = 0; jj < FBK; ++jj) Ps[tid][jj] = (__bf16)0.f;
      } else {
        c = __expf(mo - mx);
#pragma unroll
        for (int jj = 0; jj < FBK; ++jj) {
          float p = __expf(Ss[tid][jj] - mx);
          ladd += p;
          Ps[tid][jj] = (__bf16)p;
        }
      }
      row_m[tid] = mx;
      row_l[tid] = row_l[tid] * c + ladd;
      row_c[tid] = c;
    }
    __syncthreads();

    // Rescale running O, then O += P @ V.
#pragma unroll
    for (int i = 0; i < 8; ++i) {
      float c = row_c[m_base + i];
#pragma unroll
      for (int ci = 0; ci < 4; ++ci) Oacc[ci][i] *= c;
    }
    {
      v16bf a = ld_frag(&Ps[rA][kbA], &Ps[rA][16 + kbA]);
#pragma unroll
      for (int ci = 0; ci < 4; ++ci) {
        int dcol = (oci + ci) * 16 + (lane & 15);
        v16bf bb = ld_frag(&Vst[dcol][kbB], &Vst[dcol][16 + kbB]);
        Oacc[ci] = wmma_bf16(a, bb, Oacc[ci]);
      }
    }
  }
  __syncthreads();

  // Normalize and write y (bf16).
#pragma unroll
  for (int i = 0; i < 8; ++i) {
    int t = qb + m_base + i;
    float inv_l = 1.0f / row_l[m_base + i];
#pragma unroll
    for (int ci = 0; ci < 4; ++ci) {
      int d = (oci + ci) * 16 + (lane & 15);
      y[((size_t)b * T + t) * 2048 + h * 128 + d] = (__bf16)(Oacc[ci][i] * inv_l);
    }
  }
}

// ---------------------------------------------------------------------------
extern "C" void kernel_launch(void* const* d_in, const int* in_sizes, int n_in,
                              void* d_out, int out_size, void* d_ws, size_t ws_size,
                              hipStream_t stream) {
  (void)in_sizes; (void)n_in; (void)out_size; (void)ws_size;
  const float* x        = (const float*)d_in[0];
  const float* W_latent = (const float*)d_in[1];
  const float* b_latent = (const float*)d_in[2];
  const float* W_d      = (const float*)d_in[3];
  const float* b_d      = (const float*)d_in[4];
  const float* W_proj   = (const float*)d_in[5];
  const float* b_proj   = (const float*)d_in[6];
  float* out = (float*)d_out;

  const int B = 2, T = 2048, C = 2048, H = 16;
  const int M = B * T;  // 4096

  char* ws = (char*)d_ws;
  // Region map (lifetimes allow aliasing):
  //   [0, 33.5MB):  q fp32  -> later y bf16 (first 16.8MB)
  //   qrot bf16 | xb bf16 (-> later Wpt bf16) | Wdt | Wlt | kv f32 | krot | kv16
  float*  qf32 = (float*)ws;
  __bf16* yb16 = (__bf16*)ws;
  size_t off = (size_t)M * 2048 * 4;
  __bf16* qrot = (__bf16*)(ws + off);            off += (size_t)M * 2048 * 2;
  __bf16* xb   = (__bf16*)(ws + off);
  __bf16* Wpt  = (__bf16*)(ws + off);            off += (size_t)M * 2048 * 2;
  __bf16* Wdt  = (__bf16*)(ws + off);            off += (size_t)C * 2048 * 2;
  __bf16* Wlt  = (__bf16*)(ws + off);            off += (size_t)C * 128 * 2;
  float*  kvf  = (float*)(ws + off);             off += (size_t)M * 128 * 4;
  __bf16* krot = (__bf16*)(ws + off);            off += (size_t)M * 128 * 2;
  __bf16* kv16 = (__bf16*)(ws + off);

  dim3 blk(256);

  // 1) bf16 conversions / weight transposes.
  convert_bf16_kernel<<<(M * C / 8 + 255) / 256, blk, 0, stream>>>(x, xb, M * C / 8);
  transpose_to_bf16<<<dim3(128 / 32, C / 32), blk, 0, stream>>>(W_latent, Wlt, C, 128);
  transpose_to_bf16<<<dim3(2048 / 32, C / 32), blk, 0, stream>>>(W_d, Wdt, C, 2048);

  // 2) kv and q GEMMs (bf16 in, f32 out).
  gemm_bf16t_wmma<<<dim3(128 / GN, M / GM), blk, 0, stream>>>(
      xb, Wlt, b_latent, kvf, M, 128, C);
  gemm_bf16t_wmma<<<dim3(2048 / GN, M / GM), blk, 0, stream>>>(
      xb, Wdt, b_d, qf32, M, 2048, C);

  // 3) RoPE -> bf16 (k also emits unrotated bf16 V); xb now dead -> Wpt reuses it.
  rope_to_bf16<<<(M * 64 + 255) / 256, blk, 0, stream>>>(kvf, krot, kv16, M, 1, T);
  rope_to_bf16<<<(M * H * 64 + 255) / 256, blk, 0, stream>>>(
      qf32, qrot, (__bf16*)nullptr, M * H, H, T);
  transpose_to_bf16<<<dim3(2048 / 32, C / 32), blk, 0, stream>>>(W_proj, Wpt, C, 2048);

  // 4) Flash attention -> y bf16 (overwrites dead q fp32 region).
  flash_mla_wmma<<<dim3(B * H, T / FBQ), blk, 0, stream>>>(qrot, krot, kv16, yb16, B, T);

  // 5) out = y @ W_proj + b_proj.
  gemm_bf16t_wmma<<<dim3(2048 / GN, M / GM), blk, 0, stream>>>(
      yb16, Wpt, b_proj, out, M, 2048, C);
}
